// Block_19464791785620
// MI455X (gfx1250) — matrix operand order repaired and checked by reading, not compiled, not run on hardware
//
#include <hip/hip_runtime.h>

typedef _Float16 half_t;
typedef __attribute__((ext_vector_type(16))) _Float16 v16h;
typedef __attribute__((ext_vector_type(8)))  float    v8f;
typedef __attribute__((ext_vector_type(4)))  int      v4i;

#define NBATCH 128
#define NT 256
#define NCH 384
#define NH 6
#define ND 64
#define BT (NBATCH * NT)   // 32768 rows
#define NFF 1536

#if __has_builtin(__builtin_amdgcn_global_load_async_to_lds_b128)
#define HAVE_ASYNC_LDS 1
typedef __attribute__((address_space(1))) v4i gv4i;  // global int4
typedef __attribute__((address_space(3))) v4i lv4i;  // LDS int4
#endif

// Stage a contiguous block of `chunks` 16-byte chunks from global to LDS.
// Uses the gfx1250 async LDS path (ASYNCcnt) when available.
__device__ __forceinline__ void stage_to_lds(const half_t* __restrict__ src,
                                             half_t* dst, int chunks, int tid) {
#ifdef HAVE_ASYNC_LDS
  for (int i = tid; i < chunks; i += 128) {
    __builtin_amdgcn_global_load_async_to_lds_b128(
        (gv4i*)(uintptr_t)(src + (size_t)i * 8),
        (lv4i*)(unsigned)(uintptr_t)(dst + (size_t)i * 8), 0, 0);
  }
#if __has_builtin(__builtin_amdgcn_s_wait_asynccnt)
  __builtin_amdgcn_s_wait_asynccnt(0);
#else
  asm volatile("s_wait_asynccnt 0x0" ::: "memory");
#endif
#else
  const uint4* s = (const uint4*)src;
  uint4* d = (uint4*)dst;
  for (int i = tid; i < chunks; i += 128) d[i] = s[i];
#endif
}

// ---------------- WMMA fragment helpers (wave32, gfx1250 layouts) ----------

// A-matrix 16x32 f16: lanes 0-15 row M=lane, VGPR0-3 K=0..7, VGPR4-7 K=16..23;
// lanes 16-31: K=8..15 / 24..31.  rowk points at (row, kbase).
__device__ __forceinline__ v16h frag_load_A(const half_t* rowk, int lane) {
  union { v16h h; unsigned u[8]; } fr;
  const int khalf = (lane >> 4) << 3;
#pragma unroll
  for (int j = 0; j < 8; ++j) {
    const int koff = ((j >> 2) << 4) + khalf + ((j & 3) << 1);
    fr.u[j] = *(const unsigned*)(rowk + koff);
  }
  return fr.h;
}

// B-matrix 32x16 f16: lanes 0-15 col N=lane, V0-3 K=0..7, V4-7 K=16..23;
// lanes 16-31: K=8..15 / 24..31.  rowk points at (col, kbase) in a [N][K] layout.
__device__ __forceinline__ v16h frag_load_B(const half_t* rowk, int lane) {
  union { v16h h; unsigned u[8]; } fr;
  const int kb = (lane >> 4) << 3;
#pragma unroll
  for (int j = 0; j < 8; ++j)
    fr.u[j] = *(const unsigned*)(rowk + ((j >> 2) << 4) + kb + ((j & 3) << 1));
  return fr.h;
}

__device__ __forceinline__ v8f wmma32(v16h a, v16h b, v8f c) {
  return __builtin_amdgcn_wmma_f32_16x16x32_f16(false, a, false, b, (short)0, c,
                                                false, false);
}

// ---------------- weight convert + transpose: f32 [K][N] -> f16 [N][K] -----

__global__ void cvt_transpose(const float* __restrict__ in, half_t* __restrict__ out,
                              int Kd, int Nd) {
  const size_t mat = (size_t)Kd * Nd;
  const float* ip = in + (size_t)blockIdx.y * mat;
  half_t* op = out + (size_t)blockIdx.y * mat;
  const int idx = blockIdx.x * 256 + threadIdx.x;
  if (idx < (int)mat) {
    const int kk = idx / Nd, nn = idx % Nd;
    op[(size_t)nn * Kd + kk] = (half_t)ip[idx];
  }
}

// ---------------- layernorm: f32 row (C=384) -> f16 row --------------------

__global__ void ln_kernel(const float* __restrict__ x, const float* __restrict__ gamma,
                          const float* __restrict__ beta, half_t* __restrict__ out) {
  const int row = blockIdx.x;
  const int tid = threadIdx.x;            // 128 threads, 3 elems each
  const int wave = tid >> 5, lane = tid & 31;
  const float* xr = x + (size_t)row * NCH;
  float v0 = xr[tid], v1 = xr[tid + 128], v2 = xr[tid + 256];

  __shared__ float red[4];
  __shared__ float s_mean, s_inv;

  float s = v0 + v1 + v2;
#pragma unroll
  for (int m = 16; m > 0; m >>= 1) s += __shfl_xor(s, m, 32);
  if (lane == 0) red[wave] = s;
  __syncthreads();
  if (tid == 0) s_mean = (red[0] + red[1] + red[2] + red[3]) * (1.0f / NCH);
  __syncthreads();
  const float mean = s_mean;
  const float d0 = v0 - mean, d1 = v1 - mean, d2 = v2 - mean;
  float q = d0 * d0 + d1 * d1 + d2 * d2;
#pragma unroll
  for (int m = 16; m > 0; m >>= 1) q += __shfl_xor(q, m, 32);
  if (lane == 0) red[wave] = q;
  __syncthreads();
  if (tid == 0)
    s_inv = rsqrtf((red[0] + red[1] + red[2] + red[3]) * (1.0f / NCH) + 1e-5f);
  __syncthreads();
  const float inv = s_inv;
  half_t* orow = out + (size_t)row * NCH;
  orow[tid]       = (half_t)(d0 * inv * gamma[tid]       + beta[tid]);
  orow[tid + 128] = (half_t)(d1 * inv * gamma[tid + 128] + beta[tid + 128]);
  orow[tid + 256] = (half_t)(d2 * inv * gamma[tid + 256] + beta[tid + 256]);
}

// ---------------- QKV projection: h[BT,384] @ W[h][384,64] -----------------
// grid = (BT/16, NH, 3), block = 128 (4 waves, each a 16x16 output tile)

__global__ void qkv_kernel(const half_t* __restrict__ h16,
                           const half_t* __restrict__ wqT,
                           const half_t* __restrict__ wkT,
                           const half_t* __restrict__ wvT,
                           half_t* __restrict__ q16, half_t* __restrict__ k16,
                           half_t* __restrict__ vT16) {
  __shared__ __align__(16) half_t At[16 * NCH];
  const int tid = threadIdx.x;
  const int row0 = blockIdx.x * 16;
  const int head = blockIdx.y;
  const int z = blockIdx.z;
  stage_to_lds(h16 + (size_t)row0 * NCH, At, 16 * NCH / 8, tid);
  __syncthreads();
  const int wave = tid >> 5, lane = tid & 31;
  const int n0 = wave * 16;
  const half_t* wT = (z == 0 ? wqT : (z == 1 ? wkT : wvT)) + (size_t)head * ND * NCH;
  const half_t* arow = At + (lane & 15) * NCH;
  const half_t* brow = wT + (size_t)(n0 + (lane & 15)) * NCH;
  v8f acc = {};
#pragma unroll
  for (int ks = 0; ks < NCH / 32; ++ks)
    acc = wmma32(frag_load_A(arow + ks * 32, lane),
                 frag_load_B(brow + ks * 32, lane), acc);
  const int Ncol = lane & 15, Mb = (lane >> 4) << 3;
#pragma unroll
  for (int i = 0; i < 8; ++i) {
    const int row = row0 + Mb + i;
    const int b = row >> 8, t = row & (NT - 1);
    const int d = n0 + Ncol;
    const half_t val = (half_t)acc[i];
    if (z == 2)
      vT16[(((size_t)b * NH + head) * ND + d) * NT + t] = val;   // [B,H,D,T]
    else if (z == 1)
      k16[(((size_t)b * NH + head) * NT + t) * ND + d] = val;    // [B,H,T,D]
    else
      q16[(((size_t)b * NH + head) * NT + t) * ND + d] = val;
  }
}

// ---------------- causal attention per (b,h), 16 query rows per block ------
// grid = (T/16, B*H), block = 128

__global__ void attn_kernel(const half_t* __restrict__ q16,
                            const half_t* __restrict__ k16,
                            const half_t* __restrict__ vT16,
                            half_t* __restrict__ attn16) {
  __shared__ __align__(16) float S[16][NT];
  __shared__ float red[16][8];
  __shared__ float rowmax[16], rowsum[16];
  __shared__ __align__(16) half_t P[16][NT];

  const int tid = threadIdx.x;
  const int wave = tid >> 5, lane = tid & 31;
  const int m0 = blockIdx.x * 16;
  const int bh = blockIdx.y;
  const int b = bh / NH, hd = bh % NH;
  const half_t* q = q16 + (size_t)bh * NT * ND;
  const half_t* k = k16 + (size_t)bh * NT * ND;
  const half_t* vt = vT16 + (size_t)bh * ND * NT;
  const float scale = 0.05103103630798288f;  // 384^-0.5 (full C, per reference)

  const int Ncol = lane & 15, Mb = (lane >> 4) << 3;
  const v16h a0 = frag_load_A(q + (size_t)(m0 + (lane & 15)) * ND + 0, lane);
  const v16h a1 = frag_load_A(q + (size_t)(m0 + (lane & 15)) * ND + 32, lane);
#pragma unroll
  for (int st = 0; st < 4; ++st) {
    const int s0 = wave * 64 + st * 16;
    const half_t* br = k + (size_t)(s0 + (lane & 15)) * ND;
    v8f acc = {};
    acc = wmma32(a0, frag_load_B(br + 0, lane), acc);
    acc = wmma32(a1, frag_load_B(br + 32, lane), acc);
#pragma unroll
    for (int i = 0; i < 8; ++i) {
      const int r = Mb + i;
      const int col = s0 + Ncol;
      S[r][col] = (col <= m0 + r) ? acc[i] * scale : -1e30f;
    }
  }
  __syncthreads();

  // block softmax: 8 threads per row, 32 cols each
  const int r = tid >> 3, g = tid & 7;
  float mx = -1e30f;
  for (int c = 0; c < 32; ++c) mx = fmaxf(mx, S[r][g * 32 + c]);
  red[r][g] = mx;
  __syncthreads();
  if (g == 0) {
    float m = red[r][0];
    for (int j = 1; j < 8; ++j) m = fmaxf(m, red[r][j]);
    rowmax[r] = m;
  }
  __syncthreads();
  const float rm = rowmax[r];
  float sm = 0.f;
  for (int c = 0; c < 32; ++c) {
    const float e = __expf(S[r][g * 32 + c] - rm);
    S[r][g * 32 + c] = e;
    sm += e;
  }
  red[r][g] = sm;
  __syncthreads();
  if (g == 0) {
    float s = 0.f;
    for (int j = 0; j < 8; ++j) s += red[r][j];
    rowsum[r] = s;
  }
  __syncthreads();
  const float inv = 1.f / rowsum[r];
  for (int c = 0; c < 32; ++c) P[r][g * 32 + c] = (half_t)(S[r][g * 32 + c] * inv);
  __syncthreads();

  // out = P[16,256] @ V[256,64]; wave -> 16 output cols
  const int n0 = wave * 16;
  const half_t* ar = &P[lane & 15][0];
  const half_t* br = vt + (size_t)(n0 + (lane & 15)) * NT;
  v8f acc2 = {};
#pragma unroll
  for (int ks = 0; ks < NT / 32; ++ks)
    acc2 = wmma32(frag_load_A(ar + ks * 32, lane),
                  frag_load_B(br + ks * 32, lane), acc2);
#pragma unroll
  for (int i = 0; i < 8; ++i) {
    const int tq = m0 + Mb + i;
    attn16[((size_t)(b * NT + tq)) * NCH + hd * ND + n0 + Ncol] = (half_t)acc2[i];
  }
}

// ---------------- generic row-tile GEMM + epilogue -------------------------
// Y[row, n] = A[row, :K] @ Bt[n, :K] + bias[n]; EPI 0: +resid (f32 out),
// EPI 1: relu (f16 out).  grid = (BT/16, ldo/64), block = 128.

template <int K, int EPI>
__global__ void gemm_rowtile(const half_t* __restrict__ A,
                             const half_t* __restrict__ Bt,
                             const float* __restrict__ bias,
                             const float* __restrict__ resid,
                             float* __restrict__ outf, half_t* __restrict__ outh,
                             int ldo) {
  __shared__ __align__(16) half_t At[16 * K];
  const int tid = threadIdx.x;
  const int row0 = blockIdx.x * 16;
  stage_to_lds(A + (size_t)row0 * K, At, 16 * K / 8, tid);
  __syncthreads();
  const int wave = tid >> 5, lane = tid & 31;
  const int n0 = blockIdx.y * 64 + wave * 16;
  const half_t* arow = At + (lane & 15) * K;
  const half_t* brow = Bt + (size_t)(n0 + (lane & 15)) * K;
  v8f acc = {};
#pragma unroll 4
  for (int ks = 0; ks < K / 32; ++ks)
    acc = wmma32(frag_load_A(arow + ks * 32, lane),
                 frag_load_B(brow + ks * 32, lane), acc);
  const int Ncol = lane & 15, Mb = (lane >> 4) << 3;
  const int col = n0 + Ncol;
  const float bv = bias[col];
#pragma unroll
  for (int i = 0; i < 8; ++i) {
    const size_t row = (size_t)row0 + Mb + i;
    const float v = acc[i] + bv;
    if (EPI == 0)
      outf[row * ldo + col] = resid[row * ldo + col] + v;
    else
      outh[row * ldo + col] = (half_t)fmaxf(v, 0.f);
  }
}

// ---------------- host side ------------------------------------------------

extern "C" void kernel_launch(void* const* d_in, const int* in_sizes, int n_in,
                              void* d_out, int out_size, void* d_ws, size_t ws_size,
                              hipStream_t stream) {
  const float* x      = (const float*)d_in[0];
  const float* wq     = (const float*)d_in[1];
  const float* wk     = (const float*)d_in[2];
  const float* wv     = (const float*)d_in[3];
  const float* w_proj = (const float*)d_in[4];
  const float* b_proj = (const float*)d_in[5];
  const float* w1     = (const float*)d_in[6];
  const float* b1     = (const float*)d_in[7];
  const float* w2     = (const float*)d_in[8];
  const float* b2     = (const float*)d_in[9];
  const float* ln1_g  = (const float*)d_in[10];
  const float* ln1_b  = (const float*)d_in[11];
  const float* ln2_g  = (const float*)d_in[12];
  const float* ln2_b  = (const float*)d_in[13];

  size_t off = 0;
  char* base = (char*)d_ws;
  auto wsalloc = [&](size_t bytes) -> void* {
    void* p = base + off;
    off += (bytes + 255) & ~(size_t)255;
    return p;
  };
  half_t* wqT    = (half_t*)wsalloc((size_t)NH * ND * NCH * 2);
  half_t* wkT    = (half_t*)wsalloc((size_t)NH * ND * NCH * 2);
  half_t* wvT    = (half_t*)wsalloc((size_t)NH * ND * NCH * 2);
  half_t* wprojT = (half_t*)wsalloc((size_t)NCH * NCH * 2);
  half_t* w1T    = (half_t*)wsalloc((size_t)NFF * NCH * 2);
  half_t* w2T    = (half_t*)wsalloc((size_t)NCH * NFF * 2);
  half_t* h16    = (half_t*)wsalloc((size_t)BT * NCH * 2);
  half_t* q16    = (half_t*)wsalloc((size_t)BT * NCH * 2);
  half_t* k16    = (half_t*)wsalloc((size_t)BT * NCH * 2);
  half_t* vT16   = (half_t*)wsalloc((size_t)BT * NCH * 2);
  half_t* attn16 = (half_t*)wsalloc((size_t)BT * NCH * 2);
  float*  x2     = (float*)wsalloc((size_t)BT * NCH * 4);
  half_t* h2_16  = (half_t*)wsalloc((size_t)BT * NCH * 2);
  half_t* hidden = (half_t*)wsalloc((size_t)BT * NFF * 2);
  (void)ws_size; (void)n_in; (void)in_sizes; (void)out_size;

  // weights -> f16 transposed [N][K]
  cvt_transpose<<<dim3((NCH * ND + 255) / 256, NH), 256, 0, stream>>>(wq, wqT, NCH, ND);
  cvt_transpose<<<dim3((NCH * ND + 255) / 256, NH), 256, 0, stream>>>(wk, wkT, NCH, ND);
  cvt_transpose<<<dim3((NCH * ND + 255) / 256, NH), 256, 0, stream>>>(wv, wvT, NCH, ND);
  cvt_transpose<<<dim3((NCH * NCH + 255) / 256, 1), 256, 0, stream>>>(w_proj, wprojT, NCH, NCH);
  cvt_transpose<<<dim3((NCH * NFF + 255) / 256, 1), 256, 0, stream>>>(w1, w1T, NCH, NFF);
  cvt_transpose<<<dim3((NFF * NCH + 255) / 256, 1), 256, 0, stream>>>(w2, w2T, NFF, NCH);

  // LN1
  ln_kernel<<<BT, 128, 0, stream>>>(x, ln1_g, ln1_b, h16);
  // QKV
  qkv_kernel<<<dim3(BT / 16, NH, 3), 128, 0, stream>>>(h16, wqT, wkT, wvT, q16, k16, vT16);
  // attention
  attn_kernel<<<dim3(NT / 16, NBATCH * NH), 128, 0, stream>>>(q16, k16, vT16, attn16);
  // proj + residual -> x2 (f32)
  gemm_rowtile<NCH, 0><<<dim3(BT / 16, NCH / 64), 128, 0, stream>>>(
      attn16, wprojT, b_proj, x, x2, nullptr, NCH);
  // LN2
  ln_kernel<<<BT, 128, 0, stream>>>(x2, ln2_g, ln2_b, h2_16);
  // MLP1: relu(h2 @ w1 + b1) -> hidden (f16)
  gemm_rowtile<NCH, 1><<<dim3(BT / 16, NFF / 64), 128, 0, stream>>>(
      h2_16, w1T, b1, nullptr, nullptr, hidden, NFF);
  // MLP2 + residual -> out (f32)
  gemm_rowtile<NFF, 0><<<dim3(BT / 16, NCH / 64), 128, 0, stream>>>(
      hidden, w2T, b2, x2, (float*)d_out, nullptr, NCH);
}
